// PDE_M1_85770496901490
// MI455X (gfx1250) — matrix operand order repaired and checked by reading, not compiled, not run on hardware
//
#include <hip/hip_runtime.h>
#include <hip/hip_bf16.h>

typedef __attribute__((ext_vector_type(16))) _Float16 v16h;
typedef __attribute__((ext_vector_type(2)))  _Float16 h2;
typedef __attribute__((ext_vector_type(8)))  float    v8f;

#define N_MET 250000
#define N_RXN 500000
#define E_SUB 2000000
#define E_ALL 4000000
#define HIDDEN 64
#define MSG_DIM 32
#define DT 0.01f

#define NTILE_SUB (E_SUB / 16)   // 125000
#define NTILE_RXN (N_RXN / 16)   // 31250

// ---- workspace layout (bytes, all 32B aligned) ----
#define OFF_HRXN 0ul                    // N_RXN*32 f32 = 64,000,000
#define OFF_V    64000000ul             // N_RXN f32    =  2,000,000
#define OFF_TOT  66000000ul             // N_MET f32    =  1,000,000
#define OFF_MS   67000000ul             // N_MET f32    =  1,000,000
#define OFF_RS   68000000ul             // N_RXN i32    =  2,000,000
#define OFF_PW2  70000000ul             // 2048 f16     =      4,096
#define OFF_PW3  70004096ul             // 2048 f16     =      4,096
#define OFF_PW1  70008192ul             // 192 f16      =        384

static __device__ __forceinline__ v8f wmma16x16x32(v16h a, v16h b, v8f c) {
    return __builtin_amdgcn_wmma_f32_16x16x32_f16(
        /*neg_a=*/false, a, /*neg_b=*/false, b,
        /*c_mod=*/(short)0, c, /*reuse_a=*/false, /*reuse_b=*/false);
}

// ------------------------------------------------------------------
// Kernel 0: zero h_rxn / total / out, init rxn_scale bits to 1.0f
// ------------------------------------------------------------------
__global__ void k_init(float* __restrict__ h_rxn, float* __restrict__ total,
                       float* __restrict__ out, int* __restrict__ rxn_scale) {
    int i = blockIdx.x * blockDim.x + threadIdx.x;
    if (i < N_RXN * MSG_DIM) h_rxn[i] = 0.0f;
    if (i < N_MET) { total[i] = 0.0f; out[i] = 0.0f; }
    if (i < N_RXN) rxn_scale[i] = 0x3f800000; // 1.0f
}

// ------------------------------------------------------------------
// Kernel 1: pack W2 (64x32) / W3 (32x64) into this file's contiguous-k B-fragment
// layout (frag[f][lane][e], K=(lane>>4)*16+e (+32*ks), col=(lane&15)+16*t),
// and W1/b1 into flat f16 (w0[64] | w1[64] | b1[64]).
// ------------------------------------------------------------------
__global__ void k_pack(const float* __restrict__ W2, const float* __restrict__ W3,
                       const float* __restrict__ W1, const float* __restrict__ b1,
                       _Float16* __restrict__ pW2, _Float16* __restrict__ pW3,
                       _Float16* __restrict__ pW1) {
    int i = blockIdx.x * blockDim.x + threadIdx.x;  // 0..4607
    if (i < 2048) {
        int e = i & 15, lane = (i >> 4) & 31, frag = (i >> 9) & 3;
        int t = frag >> 1, ks = frag & 1;
        int k = ks * 32 + (lane >> 4) * 16 + e;
        int col = (lane & 15) + t * 16;
        pW2[frag * 512 + lane * 16 + e] = (_Float16)W2[k * MSG_DIM + col];
    } else if (i < 4096) {
        int j = i - 2048;
        int e = j & 15, lane = (j >> 4) & 31, frag = (j >> 9) & 3;
        int k = (lane >> 4) * 16 + e;
        int col = (lane & 15) + frag * 16;
        pW3[frag * 512 + lane * 16 + e] = (_Float16)W3[k * HIDDEN + col];
    } else if (i < 4096 + 192) {
        int j = i - 4096;
        float vsrc = (j < 128) ? W1[j] : b1[j - 128];
        pW1[j] = (_Float16)vsrc;
    }
}

// ------------------------------------------------------------------
// Kernel 2: edge MLP. One wave per 16-edge tile (grid-stride).
// layer1 (K=2) in VALU directly into A-fragment layout (f16 weights,
// packed 2/VGPR, hoisted: 48 regs); layer2 via 4x v_wmma_f32_16x16x32_f16;
// atomic scatter into h_rxn.
// ------------------------------------------------------------------
__global__ void __launch_bounds__(256, 1)
k_edge_msg(const float* __restrict__ x,
           const int* __restrict__ met_sub,
           const int* __restrict__ rxn_sub,
           const float* __restrict__ sto_sub,
           const _Float16* __restrict__ pW1,   // w0[64] | w1[64] | b1[64]
           const float* __restrict__ b2,       // (32)
           const _Float16* __restrict__ pW2,
           float* __restrict__ h_rxn) {
    const int lane = threadIdx.x & 31;
    const int hi = lane >> 4;
    const int row = lane & 15;
    const int wid = (blockIdx.x * blockDim.x + threadIdx.x) >> 5;
    const int nw = (gridDim.x * blockDim.x) >> 5;

    // B fragments: loop invariant, 32B contiguous per lane
    v16h B00 = *(const v16h*)(pW2 + 0 * 512 + lane * 16); // t=0, ks=0
    v16h B01 = *(const v16h*)(pW2 + 1 * 512 + lane * 16); // t=0, ks=1
    v16h B10 = *(const v16h*)(pW2 + 2 * 512 + lane * 16); // t=1, ks=0
    v16h B11 = *(const v16h*)(pW2 + 3 * 512 + lane * 16); // t=1, ks=1

    // Hoist layer-1 weights for this lane's 32 K slots, packed f16 pairs.
    // idx = 2j(+1) ; K = ((idx>>4)<<5) + hi*16 + (idx&15)
    h2 w0h[16], w1h[16], bh[16];
#pragma unroll
    for (int j = 0; j < 16; ++j) {
        int idx = 2 * j;
        int k = ((idx >> 4) << 5) + (hi << 4) + (idx & 15);
        w0h[j] = *(const h2*)(pW1 + k);
        w1h[j] = *(const h2*)(pW1 + 64 + k);
        bh[j]  = *(const h2*)(pW1 + 128 + k);
    }
    const float bias0 = b2[row];
    const float bias1 = b2[16 + row];

    for (int tile = wid; tile < NTILE_SUB; tile += nw) {
        const int tileBase = tile * 16;
        const int eid = tileBase + row;          // lanes L and L+16 share edge row
        const float c = x[(size_t)met_sub[eid] * 8 + 3];
        const float s = sto_sub[eid];

        v16h A0, A1;
#pragma unroll
        for (int idx = 0; idx < 32; ++idx) {
            const int j = idx >> 1, p = idx & 1;
            float h = tanhf(fmaf(c, (float)w0h[j][p],
                            fmaf(s, (float)w1h[j][p], (float)bh[j][p])));
            if (idx < 16) A0[idx] = (_Float16)h;
            else          A1[idx - 16] = (_Float16)h;
        }

        v8f acc0 = {}; v8f acc1 = {};
        acc0 = wmma16x16x32(A0, B00, acc0);
        acc0 = wmma16x16x32(A1, B01, acc0);
        acc1 = wmma16x16x32(A0, B10, acc1);
        acc1 = wmma16x16x32(A1, B11, acc1);

        // C layout: VGPR r -> M = r + 8*hi, N = row (+16 for acc1)
#pragma unroll
        for (int r = 0; r < 8; ++r) {
            int eid2 = tileBase + r + (hi << 3);
            int rx = rxn_sub[eid2];
            float* dst = h_rxn + (size_t)rx * MSG_DIM;
            atomicAdd(dst + row,      acc0[r] + bias0);
            atomicAdd(dst + 16 + row, acc1[r] + bias1);
        }
    }
}

// ------------------------------------------------------------------
// Kernel 3: reaction MLP. One wave per 16-rxn tile.
// (16x32) @ W3 (32x64) via 4 WMMAs, tanh, dot W4, lane reduce,
// v = 10^log_k * softplus(r)
// ------------------------------------------------------------------
__global__ void __launch_bounds__(256, 1)
k_rxn_mlp(const float* __restrict__ h_rxn,
          const _Float16* __restrict__ pW3,
          const float* __restrict__ b3,  // (64)
          const float* __restrict__ W4,  // (64)
          const float* __restrict__ b4,  // (1)
          const float* __restrict__ log_k,
          float* __restrict__ v) {
    const int lane = threadIdx.x & 31;
    const int hi = lane >> 4;
    const int row = lane & 15;
    const int wid = (blockIdx.x * blockDim.x + threadIdx.x) >> 5;
    const int nw = (gridDim.x * blockDim.x) >> 5;

    v16h B0 = *(const v16h*)(pW3 + 0 * 512 + lane * 16);
    v16h B1 = *(const v16h*)(pW3 + 1 * 512 + lane * 16);
    v16h B2 = *(const v16h*)(pW3 + 2 * 512 + lane * 16);
    v16h B3 = *(const v16h*)(pW3 + 3 * 512 + lane * 16);

    float b3v[4], w4v[4];
#pragma unroll
    for (int t = 0; t < 4; ++t) {
        b3v[t] = b3[row + 16 * t];
        w4v[t] = W4[row + 16 * t];
    }
    const float b4s = b4[0];

    for (int tile = wid; tile < NTILE_RXN; tile += nw) {
        const int tileBase = tile * 16;
        const int rid = tileBase + row;  // A-matrix: lane -> M = lane&15
        const float* hp = h_rxn + (size_t)rid * MSG_DIM;

        v16h A;
#pragma unroll
        for (int e = 0; e < 16; ++e) {
            int k = (hi << 4) + e;
            A[e] = (_Float16)hp[k];
        }

        v8f z = {};
        v8f c0 = wmma16x16x32(A, B0, z);
        v8f c1 = wmma16x16x32(A, B1, z);
        v8f c2 = wmma16x16x32(A, B2, z);
        v8f c3 = wmma16x16x32(A, B3, z);

        float acc[8];
#pragma unroll
        for (int r = 0; r < 8; ++r) {
            float a = tanhf(c0[r] + b3v[0]) * w4v[0];
            a = fmaf(tanhf(c1[r] + b3v[1]), w4v[1], a);
            a = fmaf(tanhf(c2[r] + b3v[2]), w4v[2], a);
            a = fmaf(tanhf(c3[r] + b3v[3]), w4v[3], a);
            acc[r] = a;
        }

        // Reduce across 16 lanes within each half (rows r and r+8)
#pragma unroll
        for (int r = 0; r < 8; ++r) {
            float a = acc[r];
            a += __shfl_xor(a, 1, 32);
            a += __shfl_xor(a, 2, 32);
            a += __shfl_xor(a, 4, 32);
            a += __shfl_xor(a, 8, 32);
            if (row == r) {
                int rid2 = tileBase + r + (hi << 3);
                float rr = a + b4s;
                float kk = exp2f(log_k[rid2] * 3.3219280948873623f); // 10^log_k
                float sp = fmaxf(rr, 0.0f) + logf(1.0f + expf(-fabsf(rr)));
                v[rid2] = kk * sp;
            }
        }
    }
}

// ------------------------------------------------------------------
// Epilogue kernels
// ------------------------------------------------------------------
__global__ void k_consume(const int* __restrict__ met_sub, const int* __restrict__ rxn_sub,
                          const float* __restrict__ sto_sub, const float* __restrict__ v,
                          float* __restrict__ total) {
    int e = blockIdx.x * blockDim.x + threadIdx.x;
    if (e < E_SUB) {
        float cons = sto_sub[e] * v[rxn_sub[e]] * DT;
        atomicAdd(total + met_sub[e], cons);
    }
}

__global__ void k_met_scale(const float* __restrict__ x, const float* __restrict__ total,
                            float* __restrict__ met_scale) {
    int i = blockIdx.x * blockDim.x + threadIdx.x;
    if (i < N_MET) {
        float t = total[i];
        float ms = 1.0f;
        if (t > 1e-12f) ms = fminf(x[(size_t)i * 8 + 3] / t, 1.0f);
        met_scale[i] = ms;
    }
}

__global__ void k_rxn_min(const int* __restrict__ met_sub, const int* __restrict__ rxn_sub,
                          const float* __restrict__ met_scale, int* __restrict__ rxn_scale) {
    int e = blockIdx.x * blockDim.x + threadIdx.x;
    if (e < E_SUB) {
        // scales are positive -> int ordering == float ordering
        atomicMin(rxn_scale + rxn_sub[e], __float_as_int(met_scale[met_sub[e]]));
    }
}

__global__ void k_vscale(float* __restrict__ v, const int* __restrict__ rxn_scale) {
    int i = blockIdx.x * blockDim.x + threadIdx.x;
    if (i < N_RXN) v[i] *= __int_as_float(rxn_scale[i]);
}

__global__ void k_final(const int* __restrict__ met_all, const int* __restrict__ rxn_all,
                        const float* __restrict__ sto_all, const float* __restrict__ v,
                        float* __restrict__ out) {
    int e = blockIdx.x * blockDim.x + threadIdx.x;
    if (e < E_ALL) {
        atomicAdd(out + met_all[e], sto_all[e] * v[rxn_all[e]]);
    }
}

// ------------------------------------------------------------------
extern "C" void kernel_launch(void* const* d_in, const int* in_sizes, int n_in,
                              void* d_out, int out_size, void* d_ws, size_t ws_size,
                              hipStream_t stream) {
    const float* x       = (const float*)d_in[0];
    const int*   met_sub = (const int*)d_in[1];
    const int*   rxn_sub = (const int*)d_in[2];
    const float* sto_sub = (const float*)d_in[3];
    const int*   met_all = (const int*)d_in[4];
    const int*   rxn_all = (const int*)d_in[5];
    const float* sto_all = (const float*)d_in[6];
    const float* W1      = (const float*)d_in[7];
    const float* b1      = (const float*)d_in[8];
    const float* W2      = (const float*)d_in[9];
    const float* b2      = (const float*)d_in[10];
    const float* W3      = (const float*)d_in[11];
    const float* b3      = (const float*)d_in[12];
    const float* W4      = (const float*)d_in[13];
    const float* b4      = (const float*)d_in[14];
    const float* log_k   = (const float*)d_in[15];

    char* ws = (char*)d_ws;
    float*     h_rxn     = (float*)(ws + OFF_HRXN);
    float*     v         = (float*)(ws + OFF_V);
    float*     total     = (float*)(ws + OFF_TOT);
    float*     met_scale = (float*)(ws + OFF_MS);
    int*       rxn_scale = (int*)(ws + OFF_RS);
    _Float16*  pW2       = (_Float16*)(ws + OFF_PW2);
    _Float16*  pW3       = (_Float16*)(ws + OFF_PW3);
    _Float16*  pW1       = (_Float16*)(ws + OFF_PW1);
    float*     out       = (float*)d_out;

    k_init<<<(N_RXN * MSG_DIM + 255) / 256, 256, 0, stream>>>(h_rxn, total, out, rxn_scale);
    k_pack<<<18, 256, 0, stream>>>(W2, W3, W1, b1, pW2, pW3, pW1);
    k_edge_msg<<<2048, 256, 0, stream>>>(x, met_sub, rxn_sub, sto_sub,
                                         pW1, b2, pW2, h_rxn);
    k_rxn_mlp<<<1024, 256, 0, stream>>>(h_rxn, pW3, b3, W4, b4, log_k, v);
    k_consume<<<(E_SUB + 255) / 256, 256, 0, stream>>>(met_sub, rxn_sub, sto_sub, v, total);
    k_met_scale<<<(N_MET + 255) / 256, 256, 0, stream>>>(x, total, met_scale);
    k_rxn_min<<<(E_SUB + 255) / 256, 256, 0, stream>>>(met_sub, rxn_sub, met_scale, rxn_scale);
    k_vscale<<<(N_RXN + 255) / 256, 256, 0, stream>>>(v, rxn_scale);
    k_final<<<(E_ALL + 255) / 256, 256, 0, stream>>>(met_all, rxn_all, sto_all, v, out);
}
